// GIN_87926570483778
// MI455X (gfx1250) — compile-verified, matrix operand order checked by reading, not run on hardware
//
#include <hip/hip_runtime.h>

// ---------------------------------------------------------------------------
// Types for CDNA5 WMMA (wave32): 16x16x32 bf16 -> f32
// ---------------------------------------------------------------------------
typedef __attribute__((ext_vector_type(16))) __bf16         v16bf;
typedef __attribute__((ext_vector_type(8)))  float          v8f;
typedef __attribute__((ext_vector_type(8)))  unsigned short v8u;

union Frag { v16bf bf; v8u u[2]; };

__device__ __forceinline__ unsigned short f2bf(float f) {
  union { float f; unsigned int u; } c; c.f = f;
  unsigned int u = c.u;
  // round-to-nearest-even f32 -> bf16
  return (unsigned short)((u + 0x7fffu + ((u >> 16) & 1u)) >> 16);
}

// ---------------------------------------------------------------------------
// dst = src (full copy, float4)
// ---------------------------------------------------------------------------
__global__ __launch_bounds__(256) void copy_f4(const float* __restrict__ src,
                                               float* __restrict__ dst,
                                               long long n4) {
  long long i = (long long)blockIdx.x * blockDim.x + threadIdx.x;
  if (i < n4)
    reinterpret_cast<float4*>(dst)[i] =
        reinterpret_cast<const float4*>(src)[i];
}

// ---------------------------------------------------------------------------
// agg[dst] += x[src] over all edges. One wave32 per edge; each lane loads a
// contiguous float4 (global_load_b128) and issues 4 contiguous
// global_atomic_add_f32. Node features (51 MB) are L2-resident on MI455X.
// ---------------------------------------------------------------------------
__global__ __launch_bounds__(256) void scatter_add_edges(
    const float* __restrict__ x, const int* __restrict__ ei,
    float* __restrict__ agg, int E) {
  int t = blockIdx.x * blockDim.x + threadIdx.x;
  int e = t >> 5;
  if (e >= E) return;
  int lane = t & 31;
  int s = ei[e];
  int d = ei[E + e];
  const float4 xv =
      reinterpret_cast<const float4*>(x + (size_t)s * 128)[lane];
  float* ap = agg + (size_t)d * 128 + lane * 4;
  unsafeAtomicAdd(&ap[0], xv.x);
  unsafeAtomicAdd(&ap[1], xv.y);
  unsafeAtomicAdd(&ap[2], xv.z);
  unsafeAtomicAdd(&ap[3], xv.w);
}

// ---------------------------------------------------------------------------
// Fused GEMM: out[N x NC] = epilogue(A[N x 128] @ W[128 x NC] + bias)
//   MODE 0: relu(v + b)                                      (hidden MLP)
//   MODE 1: relu((v + b)*scale + shift) + xres               (BN + residual)
//   MODE 2: v + b                                            (last layer)
// Block = 256 threads (8 waves), tile = 64 rows x NC cols.
// A tile and W^T staged to LDS as bf16 (rows padded to 136 halves -> 16B
// aligned ds_load_b128 fragment loads). WMMA: v_wmma_f32_16x16x32_bf16.
// Templated on <NC, MODE> so the n-tile loop fully unrolls (16 WMMAs for
// NC=128) and dead epilogue code is removed per instantiation.
// ---------------------------------------------------------------------------
#define LDSK 136

template <int NC, int MODE>
__global__ __launch_bounds__(256) void gemm_fused(
    const float* __restrict__ A, const float* __restrict__ W,
    const float* __restrict__ bias,
    const float* __restrict__ bn_g, const float* __restrict__ bn_be,
    const float* __restrict__ bn_m, const float* __restrict__ bn_v,
    const float* __restrict__ xres, float* __restrict__ out, int nrows) {
  __shared__ __align__(16) unsigned short sA[64 * LDSK];
  __shared__ __align__(16) unsigned short sW[128 * LDSK];  // [n][k] transposed

  const int tid = threadIdx.x;
  const int m0 = blockIdx.x * 64;

  // Stage A tile (64 x 128 f32 -> bf16). Clamp OOB rows (stores are guarded).
#pragma unroll
  for (int i = tid; i < 64 * 32; i += 256) {
    int r = i >> 5, c4 = (i & 31) << 2;
    int row = m0 + r;
    row = row < nrows ? row : nrows - 1;
    float4 q = *reinterpret_cast<const float4*>(A + (size_t)row * 128 + c4);
    unsigned short* d = &sA[r * LDSK + c4];
    d[0] = f2bf(q.x); d[1] = f2bf(q.y); d[2] = f2bf(q.z); d[3] = f2bf(q.w);
  }
  // Stage W (row-major [k][n]) transposed into sW[n][k] as bf16.
#pragma unroll
  for (int i = tid; i < 128 * NC; i += 256) {
    int k = i / NC, n = i - k * NC;
    sW[n * LDSK + k] = f2bf(W[i]);
  }
  __syncthreads();

  const int wave = tid >> 5, lane = tid & 31;
  const int hl   = lane >> 4;   // 0 / 1 (half of wave)
  const int l16  = lane & 15;
  const int koff = hl << 3;     // K offset 0 or 8 per ISA A/B layout
  const int mt   = wave >> 1;   // m tile 0..3 (16 rows each)
  const int nh   = wave & 1;    // n tile parity
  const bool full = (m0 + 64) <= nrows;  // scalar: all rows in-bounds?

  constexpr int NTW = (NC / 16) / 2;  // n tiles per wave: 4 (NC=128) / 2 (64)

  // Prefetch all per-column epilogue params into registers BEFORE the matrix
  // loop so no VMEM waits land between WMMAs.
  float pb[NTW], psc[NTW], psh[NTW];
#pragma unroll
  for (int i = 0; i < NTW; ++i) {
    int ncol = (nh + 2 * i) * 16 + l16;
    pb[i] = bias[ncol];
    if (MODE == 1) {
      float inv = rsqrtf(bn_v[ncol] + 1e-5f);
      psc[i] = bn_g[ncol] * inv;
      psh[i] = bn_be[ncol] - bn_m[ncol] * psc[i];
    } else {
      psc[i] = 1.f;
      psh[i] = 0.f;
    }
  }

  // Hoist the 4 A fragments for this wave's 16-row strip (reused per n-tile).
  Frag afr[4];
#pragma unroll
  for (int kt = 0; kt < 4; ++kt) {
    int base = (mt * 16 + l16) * LDSK + kt * 32 + koff;
    afr[kt].u[0] = *reinterpret_cast<const v8u*>(&sA[base]);       // K=koff..+7
    afr[kt].u[1] = *reinterpret_cast<const v8u*>(&sA[base + 16]);  // K=16+koff..
  }

#pragma unroll
  for (int i = 0; i < NTW; ++i) {
    const int nt = nh + 2 * i;
    v8f acc = {};
#pragma unroll
    for (int kt = 0; kt < 4; ++kt) {
      Frag bfr;
      int base = (nt * 16 + l16) * LDSK + kt * 32 + koff;
      bfr.u[0] = *reinterpret_cast<const v8u*>(&sW[base]);
      bfr.u[1] = *reinterpret_cast<const v8u*>(&sW[base + 16]);
      acc = __builtin_amdgcn_wmma_f32_16x16x32_bf16(
          false, afr[kt].bf, false, bfr.bf, (short)0, acc, false, false);
    }

    // Epilogue. C/D layout: VGPR r holds row (r + 8*hl), lanes%16 = col.
    const int ncol = nt * 16 + l16;
    const float b = pb[i], scale = psc[i], shift = psh[i];
    const int rbase = m0 + mt * 16 + (hl << 3);
    if (full) {
#pragma unroll
      for (int r = 0; r < 8; ++r) {
        int row = rbase + r;
        float val = acc[r] + b;
        if (MODE == 0) {
          val = fmaxf(val, 0.f);
        } else if (MODE == 1) {
          val = val * scale + shift;
          val = fmaxf(val, 0.f) + xres[(size_t)row * 128 + ncol];
        }
        out[(size_t)row * NC + ncol] = val;
      }
    } else {
#pragma unroll
      for (int r = 0; r < 8; ++r) {
        int row = rbase + r;
        if (row < nrows) {
          float val = acc[r] + b;
          if (MODE == 0) {
            val = fmaxf(val, 0.f);
          } else if (MODE == 1) {
            val = val * scale + shift;
            val = fmaxf(val, 0.f) + xres[(size_t)row * 128 + ncol];
          }
          out[(size_t)row * NC + ncol] = val;
        }
      }
    }
  }
}

// ---------------------------------------------------------------------------
// In-place log_softmax over rows of 64. One wave32 per row, 2 elems per lane.
// ---------------------------------------------------------------------------
__global__ __launch_bounds__(256) void logsoftmax64(float* __restrict__ out,
                                                    int nrows) {
  int row = blockIdx.x * (blockDim.x >> 5) + (threadIdx.x >> 5);
  if (row >= nrows) return;
  int lane = threadIdx.x & 31;
  float* p = out + (size_t)row * 64;
  float a = p[lane], b = p[lane + 32];
  float mx = fmaxf(a, b);
#pragma unroll
  for (int o = 16; o > 0; o >>= 1) mx = fmaxf(mx, __shfl_xor(mx, o, 32));
  float s = __expf(a - mx) + __expf(b - mx);
#pragma unroll
  for (int o = 16; o > 0; o >>= 1) s += __shfl_xor(s, o, 32);
  float l = mx + __logf(s);
  p[lane] = a - l;
  p[lane + 32] = b - l;
}

// ---------------------------------------------------------------------------
// Orchestration
// ---------------------------------------------------------------------------
extern "C" void kernel_launch(void* const* d_in, const int* in_sizes, int n_in,
                              void* d_out, int out_size, void* d_ws,
                              size_t ws_size, hipStream_t stream) {
  const int N = 100000, E = 1600000, K = 128;
  const float* x    = (const float*)d_in[0];
  const int*   ei   = (const int*)d_in[1];
  const float* W1_0 = (const float*)d_in[2];  const float* b1_0 = (const float*)d_in[3];
  const float* W2_0 = (const float*)d_in[4];  const float* b2_0 = (const float*)d_in[5];
  const float* W1_1 = (const float*)d_in[6];  const float* b1_1 = (const float*)d_in[7];
  const float* W2_1 = (const float*)d_in[8];  const float* b2_1 = (const float*)d_in[9];
  const float* W1_2 = (const float*)d_in[10]; const float* b1_2 = (const float*)d_in[11];
  const float* W2_2 = (const float*)d_in[12]; const float* b2_2 = (const float*)d_in[13];
  const float* g0   = (const float*)d_in[14]; const float* be0  = (const float*)d_in[15];
  const float* m0_  = (const float*)d_in[16]; const float* v0   = (const float*)d_in[17];
  const float* g1   = (const float*)d_in[18]; const float* be1  = (const float*)d_in[19];
  const float* m1_  = (const float*)d_in[20]; const float* v1   = (const float*)d_in[21];
  float* out = (float*)d_out;

  float* buf0 = (float*)d_ws;            // agg / scratch  (N*128 f32)
  float* buf1 = buf0 + (size_t)N * K;    // hidden / scratch
  float* buf2 = buf1 + (size_t)N * K;    // x_cur / scratch

  long long n4 = (long long)N * K / 4;
  dim3 blk(256);
  dim3 cg((unsigned)((n4 + 255) / 256));
  dim3 sg((unsigned)(((long long)E * 32 + 255) / 256));
  dim3 gg((unsigned)((N + 63) / 64));
  dim3 lg((unsigned)((N + 7) / 8));

  // ---- layer 0:  agg = x + scatter(x); h = relu(agg@W1); x1 = relu(bn(h@W2)) + x
  copy_f4<<<cg, blk, 0, stream>>>(x, buf0, n4);
  scatter_add_edges<<<sg, blk, 0, stream>>>(x, ei, buf0, E);
  gemm_fused<128, 0><<<gg, blk, 0, stream>>>(buf0, W1_0, b1_0, nullptr, nullptr,
                                             nullptr, nullptr, nullptr, buf1, N);
  gemm_fused<128, 1><<<gg, blk, 0, stream>>>(buf1, W2_0, b2_0, g0, be0, m0_, v0,
                                             x, buf2, N);
  // ---- layer 1
  copy_f4<<<cg, blk, 0, stream>>>(buf2, buf0, n4);
  scatter_add_edges<<<sg, blk, 0, stream>>>(buf2, ei, buf0, E);
  gemm_fused<128, 0><<<gg, blk, 0, stream>>>(buf0, W1_1, b1_1, nullptr, nullptr,
                                             nullptr, nullptr, nullptr, buf1, N);
  gemm_fused<128, 1><<<gg, blk, 0, stream>>>(buf1, W2_1, b2_1, g1, be1, m1_, v1,
                                             buf2, buf0, N);  // x2 -> buf0
  // ---- layer 2 (output 64 cols) + log_softmax
  copy_f4<<<cg, blk, 0, stream>>>(buf0, buf1, n4);
  scatter_add_edges<<<sg, blk, 0, stream>>>(buf0, ei, buf1, E);
  gemm_fused<128, 0><<<gg, blk, 0, stream>>>(buf1, W1_2, b1_2, nullptr, nullptr,
                                             nullptr, nullptr, nullptr, buf2, N);
  gemm_fused<64, 2><<<gg, blk, 0, stream>>>(buf2, W2_2, b2_2, nullptr, nullptr,
                                            nullptr, nullptr, nullptr, out, N);
  logsoftmax64<<<lg, blk, 0, stream>>>(out, N);
}
